// MSVA_Gen_17377437680006
// MI455X (gfx1250) — matrix operand order repaired and checked by reading, not compiled, not run on hardware
//
#include <hip/hip_runtime.h>
#include <hip/hip_bf16.h>
#include <cstddef>

typedef __attribute__((ext_vector_type(16))) __bf16 v16bf;
typedef __attribute__((ext_vector_type(8)))  float  v8f;

#define N_SEQ 4096
#define M_DIM 1024
#define AP    250     // aperture: |i-j| < AP
#define BW    512     // banded logits row width (>= 2*AP-1 = 499)

// ---------------------------------------------------------------------------
// WMMA fragment loaders (bf16, 16x16x32, per CDNA5 ISA VGPR layouts)
// A (16x32 MxK): lane r=lane&15 -> row M=r; half=lane>>4; element e ->
//   K = (e>>3)*16 + half*8 + (e&7)  => two contiguous 16B chunks per lane.
// B (32x16 KxN) from K-contiguous ("Bt", N rows of K elems): lane c -> col N=c;
//   element e -> K = (e>>3)*16 + half*8 + (e&7)  => two contiguous 16B chunks per lane.
// ---------------------------------------------------------------------------
__device__ __forceinline__ v16bf load_frag_a(const __bf16* __restrict__ A, int lda,
                                             int row0, int k0, int lane) {
  int r = lane & 15, h = lane >> 4;
  union { v16bf v; uint4 q[2]; } u;
  const __bf16* p = A + (size_t)(row0 + r) * lda + k0 + h * 8;
  u.q[0] = *reinterpret_cast<const uint4*>(p);
  u.q[1] = *reinterpret_cast<const uint4*>(p + 16);
  return u.v;
}

__device__ __forceinline__ v16bf load_frag_bt(const __bf16* __restrict__ Bt, int ldb,
                                              int col0, int k0, int lane) {
  int c = lane & 15, h = lane >> 4;
  union { v16bf v; uint4 q[2]; } u;
  const __bf16* p = Bt + (size_t)(col0 + c) * ldb + k0 + h * 8;
  u.q[0] = *reinterpret_cast<const uint4*>(p);
  u.q[1] = *reinterpret_cast<const uint4*>(p + 16);
  return u.v;
}

// ---------------------------------------------------------------------------
// Generic bf16 GEMM: C = alpha * A[MxK] @ Bt[NxK]^T, f32 accumulate.
// Wave tile 32x64: 2 A fragments + 4 B fragments -> 8 WMMAs per k-step
// (0.75 b128-pairs per WMMA). All fragments issued before the WMMA group so
// waits are progressive instead of full-drain. Block = 8 waves.
// Epilogue options: f32 out, bf16 row-major out, bf16 transposed out
// (16B packed store per lane), fused bias+ReLU.
// ---------------------------------------------------------------------------
__global__ void msva_gemm_bf16(const __bf16* __restrict__ A,
                               const __bf16* __restrict__ Bt,
                               float*  __restrict__ Cf,    // optional
                               __bf16* __restrict__ Cb,    // optional (row-major)
                               __bf16* __restrict__ CbT,   // optional (transposed, ld=M)
                               const float* __restrict__ bias, // optional -> +bias, ReLU
                               float alpha, int Mrows, int Ncols, int Kdim) {
  int lane = threadIdx.x & 31, wave = threadIdx.x >> 5;
  int row0 = blockIdx.x * 32;
  int col0 = (blockIdx.y * 8 + wave) * 64;
  if (col0 >= Ncols || row0 >= Mrows) return;   // wave-uniform

  v8f acc[2][4] = {};
  for (int k0 = 0; k0 < Kdim; k0 += 32) {
    v16bf a0 = load_frag_a(A, Kdim, row0,      k0, lane);
    v16bf a1 = load_frag_a(A, Kdim, row0 + 16, k0, lane);
    v16bf b[4];
#pragma unroll
    for (int t = 0; t < 4; ++t)
      b[t] = load_frag_bt(Bt, Kdim, col0 + 16 * t, k0, lane);
#pragma unroll
    for (int t = 0; t < 4; ++t)
      acc[0][t] = __builtin_amdgcn_wmma_f32_16x16x32_bf16(false, a0, false, b[t],
                                                          (short)0, acc[0][t], false, false);
#pragma unroll
    for (int t = 0; t < 4; ++t)
      acc[1][t] = __builtin_amdgcn_wmma_f32_16x16x32_bf16(false, a1, false, b[t],
                                                          (short)0, acc[1][t], false, false);
  }

  int c = lane & 15, h = lane >> 4;
#pragma unroll
  for (int r = 0; r < 2; ++r) {
    int rbase = row0 + r * 16;
#pragma unroll
    for (int t = 0; t < 4; ++t) {
      int col = col0 + 16 * t + c;
      union { uint4 q; __bf16 e[8]; } pk;
#pragma unroll
      for (int v = 0; v < 8; ++v) {
        int row = rbase + h * 8 + v;
        float f = acc[r][t][v] * alpha;
        if (bias) f = fmaxf(f + bias[col], 0.0f);
        if (Cf) Cf[(size_t)row * Ncols + col] = f;
        if (Cb) Cb[(size_t)row * Ncols + col] = (__bf16)f;
        pk.e[v] = (__bf16)f;
      }
      if (CbT) *reinterpret_cast<uint4*>(CbT + (size_t)col * Mrows + rbase + h * 8) = pk.q;
    }
  }
}

// ---------------------------------------------------------------------------
// Banded logits: only the 33 tile diagonals with |i-j| < AP+15 are computed.
// Q is A operand; K row-major serves directly as Bt (QK^T). Output into
// compact banded storage logitsC[i][j - jlo(i)].
// ---------------------------------------------------------------------------
__global__ void msva_logits_band(const __bf16* __restrict__ Q,
                                 const __bf16* __restrict__ Kb,
                                 float* __restrict__ logitsC) {
  int lane = threadIdx.x & 31, wave = threadIdx.x >> 5;  // block = 128 (4 waves)
  int dt = blockIdx.y * 4 + wave;
  if (dt >= 33) return;                                   // wave-uniform
  int ti = blockIdx.x, tj = ti + dt - 16;
  if (tj < 0 || tj >= N_SEQ / 16) return;                 // wave-uniform
  int row0 = ti * 16, col0 = tj * 16;

  v8f acc = {};
  for (int k0 = 0; k0 < M_DIM; k0 += 64) {   // 2 k-steps per iter: overlap loads
    v16bf a0 = load_frag_a(Q, M_DIM, row0, k0, lane);
    v16bf b0 = load_frag_bt(Kb, M_DIM, col0, k0, lane);
    v16bf a1 = load_frag_a(Q, M_DIM, row0, k0 + 32, lane);
    v16bf b1 = load_frag_bt(Kb, M_DIM, col0, k0 + 32, lane);
    acc = __builtin_amdgcn_wmma_f32_16x16x32_bf16(false, a0, false, b0,
                                                  (short)0, acc, false, false);
    acc = __builtin_amdgcn_wmma_f32_16x16x32_bf16(false, a1, false, b1,
                                                  (short)0, acc, false, false);
  }
  int c = lane & 15, h = lane >> 4;
#pragma unroll
  for (int v = 0; v < 8; ++v) {
    int i = row0 + h * 8 + v, j = col0 + c, d = j - i;
    if (d > -AP && d < AP) {
      int jlo = i - (AP - 1); if (jlo < 0) jlo = 0;
      logitsC[(size_t)i * BW + (j - jlo)] = acc[v];
    }
  }
}

// ---------------------------------------------------------------------------
// Banded softmax over row i; emits att^T as bf16 (A operand for att^T@V);
// for the last stream also emits the dense f32 att row (zeros outside band).
// ---------------------------------------------------------------------------
__global__ void msva_softmax_band(const float* __restrict__ logitsC,
                                  __bf16* __restrict__ attT,
                                  float* __restrict__ attOut) {
  __shared__ float red[4];
  int i = blockIdx.x, tid = threadIdx.x;                  // block = 128
  int jlo = i - (AP - 1); if (jlo < 0) jlo = 0;
  int jhi = i + AP;       if (jhi > N_SEQ) jhi = N_SEQ;
  int w = jhi - jlo;
  const float* lc = logitsC + (size_t)i * BW;

  float mx = -1e30f;
  for (int c = tid; c < w; c += 128) mx = fmaxf(mx, lc[c]);
  for (int o = 16; o > 0; o >>= 1) mx = fmaxf(mx, __shfl_down(mx, o));
  if ((tid & 31) == 0) red[tid >> 5] = mx;
  __syncthreads();
  mx = fmaxf(fmaxf(red[0], red[1]), fmaxf(red[2], red[3]));
  __syncthreads();

  float s = 0.f;
  for (int c = tid; c < w; c += 128) s += __expf(lc[c] - mx);
  for (int o = 16; o > 0; o >>= 1) s += __shfl_down(s, o);
  if ((tid & 31) == 0) red[tid >> 5] = s;
  __syncthreads();
  float rinv = 1.0f / (red[0] + red[1] + red[2] + red[3]);

  if (attOut) {
    for (int j = tid; j < N_SEQ; j += 128) {
      float val = 0.f;
      if (j >= jlo && j < jhi) {
        val = __expf(lc[j - jlo] - mx) * rinv;
        attT[(size_t)j * N_SEQ + i] = (__bf16)val;
      }
      attOut[(size_t)i * N_SEQ + j] = val;
    }
  } else {
    for (int c = tid; c < w; c += 128) {
      float val = __expf(lc[c] - mx) * rinv;
      attT[(size_t)(jlo + c) * N_SEQ + i] = (__bf16)val;
    }
  }
}

// ---------------------------------------------------------------------------
// Y = att^T @ V with the K-loop restricted to the band around the output
// tile. A-fragment elements outside |k - j| < AP are zero-masked (handles
// workspace poison without changing EXEC around the WMMA).
// VT is V transposed [M_DIM][N_SEQ] -> contiguous B fragments.
// ---------------------------------------------------------------------------
__global__ void msva_attv_band(const __bf16* __restrict__ attT,
                               const __bf16* __restrict__ VT,
                               __bf16* __restrict__ Yb) {
  int lane = threadIdx.x & 31, wave = threadIdx.x >> 5;   // block = 256
  int j0 = blockIdx.x * 16;
  int n0 = (blockIdx.y * 8 + wave) * 64;
  int r = lane & 15, h = lane >> 4;
  int jr = j0 + r;

  v8f acc[4] = {};
  int ks = j0 - (AP - 1); if (ks < 0) ks = 0; ks &= ~31;
  int ke = j0 + 15 + AP;  if (ke > N_SEQ) ke = N_SEQ;
  for (int k0 = ks; k0 < ke; k0 += 32) {
    union { v16bf v; __bf16 e[16]; uint4 q[2]; } ua;
    const __bf16* p = attT + (size_t)jr * N_SEQ + k0 + h * 8;
    ua.q[0] = *reinterpret_cast<const uint4*>(p);
    ua.q[1] = *reinterpret_cast<const uint4*>(p + 16);
    v16bf b[4];
#pragma unroll
    for (int t = 0; t < 4; ++t)
      b[t] = load_frag_bt(VT, N_SEQ, n0 + 16 * t, k0, lane);
#pragma unroll
    for (int e = 0; e < 16; ++e) {
      int k = k0 + ((e >> 3) << 4) + h * 8 + (e & 7);
      int d = k - jr;
      bool inband = (d > -AP) && (d < AP);
      ua.e[e] = inband ? ua.e[e] : (__bf16)0.0f;
    }
#pragma unroll
    for (int t = 0; t < 4; ++t)
      acc[t] = __builtin_amdgcn_wmma_f32_16x16x32_bf16(false, ua.v, false, b[t],
                                                       (short)0, acc[t], false, false);
  }
  int c = lane & 15;
#pragma unroll
  for (int t = 0; t < 4; ++t)
#pragma unroll
    for (int v = 0; v < 8; ++v) {
      int row = j0 + h * 8 + v, col = n0 + 16 * t + c;
      Yb[(size_t)row * M_DIM + col] = (__bf16)acc[t][v];
    }
}

// ---------------------------------------------------------------------------
// LayerNorm with optional residual; unbiased variance (Bessel), matches
// gamma*(x-mean)/(sqrt(var)+eps)+beta. Optionally accumulates (stream sum).
// ---------------------------------------------------------------------------
__global__ void msva_layernorm(const float* __restrict__ A,
                               const float* __restrict__ resid,
                               const float* __restrict__ gamma,
                               const float* __restrict__ beta,
                               float* __restrict__ out, int accumulate) {
  __shared__ float red[8];
  int i = blockIdx.x, tid = threadIdx.x;                  // block = 256
  const float* a = A + (size_t)i * M_DIM;
  float vloc[4];
#pragma unroll
  for (int u = 0; u < 4; ++u) {
    int j = tid + u * 256;
    vloc[u] = a[j] + (resid ? resid[(size_t)i * M_DIM + j] : 0.0f);
  }
  float s = vloc[0] + vloc[1] + vloc[2] + vloc[3];
  for (int o = 16; o > 0; o >>= 1) s += __shfl_down(s, o);
  if ((tid & 31) == 0) red[tid >> 5] = s;
  __syncthreads();
  float mean = 0.f;
#pragma unroll
  for (int u = 0; u < 8; ++u) mean += red[u];
  mean *= (1.0f / M_DIM);
  __syncthreads();

  float sq = 0.f;
#pragma unroll
  for (int u = 0; u < 4; ++u) { float d = vloc[u] - mean; sq += d * d; }
  for (int o = 16; o > 0; o >>= 1) sq += __shfl_down(sq, o);
  if ((tid & 31) == 0) red[tid >> 5] = sq;
  __syncthreads();
  float var = 0.f;
#pragma unroll
  for (int u = 0; u < 8; ++u) var += red[u];
  var *= (1.0f / (M_DIM - 1));
  float denom = sqrtf(var) + 1e-6f;

#pragma unroll
  for (int u = 0; u < 4; ++u) {
    int j = tid + u * 256;
    float val = gamma[j] * (vloc[u] - mean) / denom + beta[j];
    size_t idx = (size_t)i * M_DIM + j;
    if (accumulate) out[idx] += val; else out[idx] = val;
  }
}

// sigmoid(h @ Wkd + bkd): one block per row.
__global__ void msva_head(const float* __restrict__ H, const float* __restrict__ Wkd,
                          const float* __restrict__ bkd, float* __restrict__ out) {
  __shared__ float red[8];
  int i = blockIdx.x, tid = threadIdx.x;                  // block = 256
  const float* h = H + (size_t)i * M_DIM;
  float s = 0.f;
#pragma unroll
  for (int u = 0; u < 4; ++u) { int j = tid + u * 256; s += h[j] * Wkd[j]; }
  for (int o = 16; o > 0; o >>= 1) s += __shfl_down(s, o);
  if ((tid & 31) == 0) red[tid >> 5] = s;
  __syncthreads();
  if (tid == 0) {
    float t = 0.f;
#pragma unroll
    for (int u = 0; u < 8; ++u) t += red[u];
    t += bkd[0];
    out[i] = 1.0f / (1.0f + __expf(-t));
  }
}

__global__ void msva_cvt_bf16(const float* __restrict__ src, __bf16* __restrict__ dst, int count) {
  int idx = blockIdx.x * 256 + threadIdx.x;
  if (idx < count) dst[idx] = (__bf16)src[idx];
}

// Weight convert + transpose: WT[n][k] = bf16(W[k][n]) (1024x1024)
__global__ void msva_cvtT_bf16(const float* __restrict__ W, __bf16* __restrict__ WT) {
  int idx = blockIdx.x * 256 + threadIdx.x;
  int n = idx >> 10, k = idx & 1023;
  WT[idx] = (__bf16)W[(size_t)k * M_DIM + n];
}

// ---------------------------------------------------------------------------
extern "C" void kernel_launch(void* const* d_in, const int* in_sizes, int n_in,
                              void* d_out, int out_size, void* d_ws, size_t ws_size,
                              hipStream_t stream) {
  (void)in_sizes; (void)n_in; (void)out_size; (void)ws_size;
  const float* x[3] = { (const float*)d_in[0], (const float*)d_in[1], (const float*)d_in[2] };
  const float* Wq  = (const float*)d_in[4];
  const float* Wk  = (const float*)d_in[5];
  const float* Wv  = (const float*)d_in[6];
  const float* Wo  = (const float*)d_in[7];
  const float* gy  = (const float*)d_in[8];
  const float* by  = (const float*)d_in[9];
  const float* Wka = (const float*)d_in[10];
  const float* bka = (const float*)d_in[11];
  const float* gka = (const float*)d_in[12];
  const float* bka2= (const float*)d_in[13];
  const float* Wkd = (const float*)d_in[14];
  const float* bkd = (const float*)d_in[15];

  char* ws = (char*)d_ws; size_t off = 0;
  auto alloc = [&](size_t bytes) -> void* {
    void* p = ws + off; off += (bytes + 255) & ~(size_t)255; return p;
  };
  const size_t WSZ = (size_t)M_DIM * M_DIM * 2;
  const size_t XB  = (size_t)N_SEQ * M_DIM * 2;
  const size_t XF  = (size_t)N_SEQ * M_DIM * 4;
  __bf16* WqT  = (__bf16*)alloc(WSZ);
  __bf16* WkT  = (__bf16*)alloc(WSZ);
  __bf16* WvT  = (__bf16*)alloc(WSZ);
  __bf16* WoT  = (__bf16*)alloc(WSZ);
  __bf16* WkaT = (__bf16*)alloc(WSZ);
  __bf16* xbf  = (__bf16*)alloc(XB);
  __bf16* Qbf  = (__bf16*)alloc(XB);
  __bf16* Kbf  = (__bf16*)alloc(XB);
  __bf16* VT   = (__bf16*)alloc(XB);
  float*  logC = (float*) alloc((size_t)N_SEQ * BW * 4);
  __bf16* attT = (__bf16*)alloc((size_t)N_SEQ * N_SEQ * 2);
  __bf16* Yb   = (__bf16*)alloc(XB);
  float*  yo   = (float*) alloc(XF);
  float*  zsum = (float*) alloc(XF);
  __bf16* zsbf = (__bf16*)alloc(XB);
  float*  hbuf = (float*) alloc(XF);
  float*  hln  = (float*) alloc(XF);

  float* outVec = (float*)d_out;
  float* outAtt = outVec + N_SEQ;

  const dim3 b256(256), b128(128);
  const dim3 gW(M_DIM * M_DIM / 256);
  const dim3 gCvt(N_SEQ * M_DIM / 256);
  const dim3 gGemm(N_SEQ / 32, M_DIM / 512);   // wave tile 32x64, 8 waves/block
  const dim3 gAttv(N_SEQ / 16, M_DIM / 512);   // wave tile 16x64 (banded K)
  const dim3 gLog(N_SEQ / 16, 9);              // 33 tile-diagonals / 4 waves
  const dim3 gRow(N_SEQ);

  msva_cvtT_bf16<<<gW, b256, 0, stream>>>(Wq,  WqT);
  msva_cvtT_bf16<<<gW, b256, 0, stream>>>(Wk,  WkT);
  msva_cvtT_bf16<<<gW, b256, 0, stream>>>(Wv,  WvT);
  msva_cvtT_bf16<<<gW, b256, 0, stream>>>(Wo,  WoT);
  msva_cvtT_bf16<<<gW, b256, 0, stream>>>(Wka, WkaT);

  for (int s = 0; s < 3; ++s) {
    msva_cvt_bf16<<<gCvt, b256, 0, stream>>>(x[s], xbf, N_SEQ * M_DIM);
    // Q = 0.06*(x@Wq), K = x@Wk (both bf16 row-major), V emitted transposed
    msva_gemm_bf16<<<gGemm, b256, 0, stream>>>(xbf, WqT, nullptr, Qbf, nullptr, nullptr,
                                               0.06f, N_SEQ, M_DIM, M_DIM);
    msva_gemm_bf16<<<gGemm, b256, 0, stream>>>(xbf, WkT, nullptr, Kbf, nullptr, nullptr,
                                               1.0f, N_SEQ, M_DIM, M_DIM);
    msva_gemm_bf16<<<gGemm, b256, 0, stream>>>(xbf, WvT, nullptr, nullptr, VT, nullptr,
                                               1.0f, N_SEQ, M_DIM, M_DIM);
    msva_logits_band<<<gLog, b128, 0, stream>>>(Qbf, Kbf, logC);
    msva_softmax_band<<<gRow, b128, 0, stream>>>(logC, attT, (s == 2) ? outAtt : nullptr);
    msva_attv_band<<<gAttv, b256, 0, stream>>>(attT, VT, Yb);
    msva_gemm_bf16<<<gGemm, b256, 0, stream>>>(Yb, WoT, yo, nullptr, nullptr, nullptr,
                                               1.0f, N_SEQ, M_DIM, M_DIM);
    msva_layernorm<<<gRow, b256, 0, stream>>>(yo, x[s], gy, by, zsum, (s == 0) ? 0 : 1);
  }

  msva_cvt_bf16<<<gCvt, b256, 0, stream>>>(zsum, zsbf, N_SEQ * M_DIM);
  msva_gemm_bf16<<<gGemm, b256, 0, stream>>>(zsbf, WkaT, hbuf, nullptr, nullptr, bka,
                                             1.0f, N_SEQ, M_DIM, M_DIM);  // +bias, ReLU
  msva_layernorm<<<gRow, b256, 0, stream>>>(hbuf, nullptr, gka, bka2, hln, 0);
  msva_head<<<gRow, b256, 0, stream>>>(hln, Wkd, bkd, outVec);
}
